// MoE_67242007986679
// MI455X (gfx1250) — compile-verified, matrix operand order checked by reading, not run on hardware
//
#include <hip/hip_runtime.h>

// ---------------------------------------------------------------------------
// MoE (top-2 of 8 experts + shared expert) for MI455X / gfx1250.
//
// router -> per-expert token gather (exactly N*TOP_K = 8192 routed slots)
// -> bf16 WMMA gate/up GEMM fused with SiLU (ping-pong LDS, register-staged
// global prefetch) -> bf16 WMMA down-projection whose A tiles stream in via
// GLOBAL_LOAD_ASYNC_TO_LDS_B128 (ASYNCcnt).  fp32 -> bf16 conversion happens
// while staging tiles (v_cvt_pk_bf16_f32); accumulate f32 with
// v_wmma_f32_16x16x32_bf16.
// ---------------------------------------------------------------------------

#define NTOK 4096
#define DH   2048   // D_HIDDEN
#define DE   1024   // D_EXPERT
#define NEXP 8
#define TOPK 2
#define SLOTS (NTOK * TOPK)

typedef __attribute__((ext_vector_type(16))) __bf16        v16bf;
typedef __attribute__((ext_vector_type(2)))  __bf16        v2bf;
typedef __attribute__((ext_vector_type(8)))  float         v8f;
typedef __attribute__((ext_vector_type(4)))  unsigned int  v4u;
typedef __attribute__((ext_vector_type(4)))  int           v4i;

union FragBF { v16bf v; v4u q[2]; };

#if defined(__HIP_DEVICE_COMPILE__) && \
    __has_builtin(__builtin_amdgcn_global_load_async_to_lds_b128)
#define HAVE_ASYNC_LDS 1
#else
#define HAVE_ASYNC_LDS 0
#endif

#if HAVE_ASYNC_LDS
__device__ __forceinline__ void async_copy_b128(const void* g, void* l) {
  __builtin_amdgcn_global_load_async_to_lds_b128(
      (__attribute__((address_space(1))) v4i*)g,
      (__attribute__((address_space(3))) v4i*)l, 0, 0);
}
__device__ __forceinline__ void wait_async0() {
#if __has_builtin(__builtin_amdgcn_s_wait_asynccnt)
  __builtin_amdgcn_s_wait_asynccnt(0);
#else
  asm volatile("s_wait_asynccnt 0x0" ::: "memory");
#endif
}
#endif

__device__ __forceinline__ float silu_f(float x) {
  return x / (1.0f + __expf(-x));
}

__device__ __forceinline__ unsigned short f2bf_bits(float f) {
  unsigned int u = __float_as_uint(f);
  u += 0x7FFFu + ((u >> 16) & 1u);
  return (unsigned short)(u >> 16);
}

// ---------------------------------------------------------------------------
// Workspace layout (bytes), total ~25.4 MB.
// ctrl ints: [0..7]=counts, [8..16]=offsets (exclusive scan), [17..24]=cursors
// ---------------------------------------------------------------------------
static constexpr size_t WS_COMBINE = 0;
static constexpr size_t WS_TOP2    = WS_COMBINE + (size_t)NTOK * NEXP * 4;
static constexpr size_t WS_CTRL    = WS_TOP2 + (size_t)NTOK * TOPK * 4;
static constexpr size_t WS_TOKIDS  = WS_CTRL + 128;
static constexpr size_t WS_HROUTE  = (WS_TOKIDS + (size_t)SLOTS * 4 + 255) & ~(size_t)255;
static constexpr size_t WS_HSHARED = WS_HROUTE + (size_t)SLOTS * DE * 2;

__global__ void moe_ctrl_init(int* __restrict__ ctrl) {
  if (threadIdx.x < 32) ctrl[threadIdx.x] = 0;
}

// ---------------------------------------------------------------------------
// 1) router
// ---------------------------------------------------------------------------
__global__ __launch_bounds__(256)
void moe_router(const float* __restrict__ X, const float* __restrict__ Wg,
                float* __restrict__ combine, int* __restrict__ top2,
                int* __restrict__ ctrl) {
  int n = blockIdx.x * blockDim.x + threadIdx.x;
  if (n >= NTOK) return;

  float l[NEXP];
#pragma unroll
  for (int e = 0; e < NEXP; ++e) l[e] = 0.0f;

  const float* xr = X + (size_t)n * DH;
  for (int d = 0; d < DH; d += 4) {
    float4 xv = *(const float4*)(xr + d);
    const float* w = Wg + (size_t)d * NEXP;
#pragma unroll
    for (int e = 0; e < NEXP; ++e)
      l[e] += xv.x * w[e] + xv.y * w[NEXP + e] + xv.z * w[2 * NEXP + e] + xv.w * w[3 * NEXP + e];
  }

  float m = l[0];
#pragma unroll
  for (int e = 1; e < NEXP; ++e) m = fmaxf(m, l[e]);
  float p[NEXP], s = 0.0f;
#pragma unroll
  for (int e = 0; e < NEXP; ++e) { p[e] = __expf(l[e] - m); s += p[e]; }
  float inv = 1.0f / s;
#pragma unroll
  for (int e = 0; e < NEXP; ++e) p[e] *= inv;

  int i1 = 0; float b1 = p[0];
#pragma unroll
  for (int e = 1; e < NEXP; ++e) if (p[e] > b1) { b1 = p[e]; i1 = e; }
  int i2 = (i1 == 0) ? 1 : 0; float b2 = p[i2];
#pragma unroll
  for (int e = 0; e < NEXP; ++e) if (e != i1 && p[e] > b2) { b2 = p[e]; i2 = e; }

#pragma unroll
  for (int e = 0; e < NEXP; ++e)
    combine[(size_t)n * NEXP + e] = (e == i1) ? b1 : ((e == i2) ? b2 : 0.0f);

  top2[n * TOPK + 0] = i1;
  top2[n * TOPK + 1] = i2;
  atomicAdd(&ctrl[i1], 1);
  atomicAdd(&ctrl[i2], 1);
}

__global__ void moe_prefix(int* __restrict__ ctrl) {
  if (threadIdx.x == 0) {
    int off = 0;
    ctrl[8] = 0;
#pragma unroll
    for (int e = 0; e < NEXP; ++e) { off += ctrl[e]; ctrl[9 + e] = off; }
  }
  if (threadIdx.x < NEXP) ctrl[17 + threadIdx.x] = 0;
}

__global__ __launch_bounds__(256)
void moe_fill(const int* __restrict__ top2, int* __restrict__ ctrl,
              int* __restrict__ tokids) {
  int n = blockIdx.x * blockDim.x + threadIdx.x;
  if (n >= NTOK) return;
#pragma unroll
  for (int k = 0; k < TOPK; ++k) {
    int e = top2[n * TOPK + k];
    int p = atomicAdd(&ctrl[17 + e], 1);
    tokids[ctrl[8 + e] + p] = n;
  }
}

// ---------------------------------------------------------------------------
// 4) fused gate/up GEMM + SiLU.  Block tile 128x64, K step 32, 8 waves each
//    own a 16x64 strip -> 8 WMMA per wave per k-step.  Ping-pong LDS with
//    register-staged global prefetch: loads for tile k+1 issue before the
//    WMMAs of tile k, convert+store after, one barrier per iteration.
// ---------------------------------------------------------------------------
template <bool ROUTED>
__global__ __launch_bounds__(256)
void moe_gateup(const float* __restrict__ X,
                const float* __restrict__ WgBase,
                const float* __restrict__ WuBase,
                const float* __restrict__ combine,
                const int* __restrict__ tokids,
                const int* __restrict__ ctrl,
                unsigned short* __restrict__ Hout) {
  const int e = blockIdx.z;
  int cnt, segOff;
  if (ROUTED) { cnt = ctrl[e]; segOff = ctrl[8 + e]; }
  else        { cnt = NTOK;    segOff = 0; }
  const int mBase = blockIdx.y * 128;
  if (mBase >= cnt) return;
  const int nBase = blockIdx.x * 64;

  const float* Wg = WgBase + (ROUTED ? (size_t)e * DH * DE : 0);
  const float* Wu = WuBase + (ROUTED ? (size_t)e * DH * DE : 0);

  __shared__ __bf16 As[2][128][32];
  __shared__ __bf16 Bg[2][64][32];
  __shared__ __bf16 Bu[2][64][32];
  __shared__ int    sTok[128];
  __shared__ float  sScale[128];

  const int tid = threadIdx.x;
  if (tid < 128) {
    int r = mBase + tid;
    int tok; float sc;
    if (ROUTED) {
      tok = (r < cnt) ? tokids[segOff + r] : 0;
      sc  = combine[(size_t)tok * NEXP + e];
    } else {
      tok = (r < NTOK) ? r : 0;
      sc  = 1.0f;
    }
    sTok[tid]   = tok;
    sScale[tid] = sc;
  }
  __syncthreads();

  const int lane = tid & 31;
  const int w    = tid >> 5;
  const int rA   = (w << 4) + (lane & 15);
  const int kc   = (lane & 16) >> 1;   // 0 or 8 (A/B fragment K-phase per lane half)

  // Per-thread source pointers, hoisted out of the k-loop.
  int   aRow[4];
  const float* aPtr[4];
#pragma unroll
  for (int i = 0; i < 4; ++i) {
    int idx = tid + i * 256;
    aRow[i] = idx >> 3;
    int c   = (idx & 7) << 2;
    aPtr[i] = X + (size_t)sTok[aRow[i]] * DH + c;
  }
  int bKr[2], bC[2];
  const float* gPtr[2];
  const float* uPtr[2];
#pragma unroll
  for (int i = 0; i < 2; ++i) {
    int idx = tid + i * 256;
    bKr[i] = idx >> 4;
    bC[i]  = (idx & 15) << 2;
    gPtr[i] = Wg + (size_t)bKr[i] * DE + nBase + bC[i];
    uPtr[i] = Wu + (size_t)bKr[i] * DE + nBase + bC[i];
  }

  float4 rA4[4], rG[2], rU[2];

  // ---- prologue: stage tile k0=0 into buffer 0 ----
#pragma unroll
  for (int i = 0; i < 4; ++i) rA4[i] = *(const float4*)(aPtr[i]);
#pragma unroll
  for (int i = 0; i < 2; ++i) { rG[i] = *(const float4*)(gPtr[i]); rU[i] = *(const float4*)(uPtr[i]); }
#pragma unroll
  for (int i = 0; i < 4; ++i) {
    int c = ((tid + i * 256) & 7) << 2;
    v2bf* d = (v2bf*)&As[0][aRow[i]][c];
    v2bf p0 = {(__bf16)rA4[i].x, (__bf16)rA4[i].y};
    v2bf p1 = {(__bf16)rA4[i].z, (__bf16)rA4[i].w};
    d[0] = p0; d[1] = p1;
  }
#pragma unroll
  for (int i = 0; i < 2; ++i) {
    Bg[0][bC[i] + 0][bKr[i]] = (__bf16)rG[i].x; Bg[0][bC[i] + 1][bKr[i]] = (__bf16)rG[i].y;
    Bg[0][bC[i] + 2][bKr[i]] = (__bf16)rG[i].z; Bg[0][bC[i] + 3][bKr[i]] = (__bf16)rG[i].w;
    Bu[0][bC[i] + 0][bKr[i]] = (__bf16)rU[i].x; Bu[0][bC[i] + 1][bKr[i]] = (__bf16)rU[i].y;
    Bu[0][bC[i] + 2][bKr[i]] = (__bf16)rU[i].z; Bu[0][bC[i] + 3][bKr[i]] = (__bf16)rU[i].w;
  }
  __syncthreads();

  v8f zero = {};
  v8f accG[4], accU[4];
#pragma unroll
  for (int t = 0; t < 4; ++t) { accG[t] = zero; accU[t] = zero; }

  int buf = 0;
  for (int k0 = 0; k0 < DH; k0 += 32) {
    const int kn = k0 + 32;
    const bool hasNext = kn < DH;

    if (hasNext) {
      // Issue next tile's loads early; s_wait lands after the WMMA block.
#pragma unroll
      for (int i = 0; i < 4; ++i) rA4[i] = *(const float4*)(aPtr[i] + kn);
#pragma unroll
      for (int i = 0; i < 2; ++i) {
        rG[i] = *(const float4*)(gPtr[i] + (size_t)kn * DE);
        rU[i] = *(const float4*)(uPtr[i] + (size_t)kn * DE);
      }
      __builtin_prefetch(gPtr[0] + (size_t)(kn + 32) * DE, 0, 1);
      __builtin_prefetch(uPtr[0] + (size_t)(kn + 32) * DE, 0, 1);
    }

    FragBF a;
    a.q[0] = *(const v4u*)&As[buf][rA][kc];
    a.q[1] = *(const v4u*)&As[buf][rA][kc + 16];
#pragma unroll
    for (int t = 0; t < 4; ++t) {
      int nb = (t << 4) + (lane & 15);
      FragBF bg, bu;
      bg.q[0] = *(const v4u*)&Bg[buf][nb][kc];
      bg.q[1] = *(const v4u*)&Bg[buf][nb][kc + 16];
      bu.q[0] = *(const v4u*)&Bu[buf][nb][kc];
      bu.q[1] = *(const v4u*)&Bu[buf][nb][kc + 16];
      accG[t] = __builtin_amdgcn_wmma_f32_16x16x32_bf16(false, a.v, false, bg.v,
                                                        (short)0, accG[t], false, false);
      accU[t] = __builtin_amdgcn_wmma_f32_16x16x32_bf16(false, a.v, false, bu.v,
                                                        (short)0, accU[t], false, false);
    }

    if (hasNext) {
      const int nb2 = buf ^ 1;
#pragma unroll
      for (int i = 0; i < 4; ++i) {
        int c = ((tid + i * 256) & 7) << 2;
        v2bf* d = (v2bf*)&As[nb2][aRow[i]][c];
        v2bf p0 = {(__bf16)rA4[i].x, (__bf16)rA4[i].y};
        v2bf p1 = {(__bf16)rA4[i].z, (__bf16)rA4[i].w};
        d[0] = p0; d[1] = p1;
      }
#pragma unroll
      for (int i = 0; i < 2; ++i) {
        Bg[nb2][bC[i] + 0][bKr[i]] = (__bf16)rG[i].x; Bg[nb2][bC[i] + 1][bKr[i]] = (__bf16)rG[i].y;
        Bg[nb2][bC[i] + 2][bKr[i]] = (__bf16)rG[i].z; Bg[nb2][bC[i] + 3][bKr[i]] = (__bf16)rG[i].w;
        Bu[nb2][bC[i] + 0][bKr[i]] = (__bf16)rU[i].x; Bu[nb2][bC[i] + 1][bKr[i]] = (__bf16)rU[i].y;
        Bu[nb2][bC[i] + 2][bKr[i]] = (__bf16)rU[i].z; Bu[nb2][bC[i] + 3][bKr[i]] = (__bf16)rU[i].w;
      }
    }
    __syncthreads();
    buf ^= 1;
  }

  // Epilogue: h = silu(gate) * up * combine, store bf16.
#pragma unroll
  for (int t = 0; t < 4; ++t) {
#pragma unroll
    for (int i = 0; i < 8; ++i) {
      int rloc = (w << 4) + i + ((lane & 16) >> 1);
      if (mBase + rloc < cnt) {
        int col = nBase + (t << 4) + (lane & 15);
        float h = silu_f(accG[t][i]) * accU[t][i] * sScale[rloc];
        Hout[(size_t)(segOff + mBase + rloc) * DE + col] = f2bf_bits(h);
      }
    }
  }
}

// ---------------------------------------------------------------------------
// 5) down-projection GEMM.  A (bf16, already in workspace) streams into LDS
//    via GLOBAL_LOAD_ASYNC_TO_LDS_B128; B (fp32 weights) is register-staged
//    + converted.  A-tile rows past `cnt` read valid scratch and only feed
//    masked-out C rows, so the copy needs no predication.
// ---------------------------------------------------------------------------
template <bool ROUTED>
__global__ __launch_bounds__(256)
void moe_down(const unsigned short* __restrict__ Hsrc,
              const float* __restrict__ WdBase,
              const int* __restrict__ tokids,
              const int* __restrict__ ctrl,
              float* __restrict__ out) {
  const int e = blockIdx.z;
  int cnt, segOff;
  if (ROUTED) { cnt = ctrl[e]; segOff = ctrl[8 + e]; }
  else        { cnt = NTOK;    segOff = 0; }
  const int mBase = blockIdx.y * 128;
  if (mBase >= cnt) return;
  const int nBase = blockIdx.x * 64;

  const float* Wd = WdBase + (ROUTED ? (size_t)e * DE * DH : 0);

  __shared__ __bf16 As[2][128][32];
  __shared__ __bf16 Bt[2][64][32];
  __shared__ int    sTok[128];

  const int tid = threadIdx.x;
  if (ROUTED && tid < 128) {
    int r = mBase + tid;
    sTok[tid] = (r < cnt) ? tokids[segOff + r] : 0;
  }

  const int lane = tid & 31;
  const int w    = tid >> 5;
  const int rA   = (w << 4) + (lane & 15);
  const int kc   = (lane & 16) >> 1;

  // A tile: 128 rows x 32 bf16 = 512 x 16B chunks; 2 per thread.
  int aRow[2], aC[2];
  const unsigned short* aPtr[2];
#pragma unroll
  for (int i = 0; i < 2; ++i) {
    int idx = tid + i * 256;
    aRow[i] = idx >> 2;
    aC[i]   = (idx & 3) << 3;
    aPtr[i] = Hsrc + (size_t)(segOff + mBase + aRow[i]) * DE + aC[i];
  }
  // B tile: 32x64 fp32, 2 float4 per thread.
  int bKr[2], bC[2];
  const float* bPtr[2];
#pragma unroll
  for (int i = 0; i < 2; ++i) {
    int idx = tid + i * 256;
    bKr[i] = idx >> 4;
    bC[i]  = (idx & 15) << 2;
    bPtr[i] = Wd + (size_t)bKr[i] * DH + nBase + bC[i];
  }

  float4 rB[2];

  // ---- prologue: tile k0=0 into buffer 0 ----
#if HAVE_ASYNC_LDS
#pragma unroll
  for (int i = 0; i < 2; ++i)
    async_copy_b128((const void*)aPtr[i], (void*)&As[0][aRow[i]][aC[i]]);
#else
#pragma unroll
  for (int i = 0; i < 2; ++i)
    *(v4u*)&As[0][aRow[i]][aC[i]] = *(const v4u*)aPtr[i];
#endif
#pragma unroll
  for (int i = 0; i < 2; ++i) rB[i] = *(const float4*)(bPtr[i]);
#pragma unroll
  for (int i = 0; i < 2; ++i) {
    Bt[0][bC[i] + 0][bKr[i]] = (__bf16)rB[i].x; Bt[0][bC[i] + 1][bKr[i]] = (__bf16)rB[i].y;
    Bt[0][bC[i] + 2][bKr[i]] = (__bf16)rB[i].z; Bt[0][bC[i] + 3][bKr[i]] = (__bf16)rB[i].w;
  }
#if HAVE_ASYNC_LDS
  wait_async0();
#endif
  __syncthreads();

  v8f zero = {};
  v8f acc[4];
#pragma unroll
  for (int t = 0; t < 4; ++t) acc[t] = zero;

  int buf = 0;
  for (int k0 = 0; k0 < DE; k0 += 32) {
    const int kn = k0 + 32;
    const bool hasNext = kn < DE;

    if (hasNext) {
      const int nb2 = buf ^ 1;
#if HAVE_ASYNC_LDS
#pragma unroll
      for (int i = 0; i < 2; ++i)
        async_copy_b128((const void*)(aPtr[i] + kn), (void*)&As[nb2][aRow[i]][aC[i]]);
#else
#pragma unroll
      for (int i = 0; i < 2; ++i)
        *(v4u*)&As[nb2][aRow[i]][aC[i]] = *(const v4u*)(aPtr[i] + kn);
#endif
#pragma unroll
      for (int i = 0; i < 2; ++i) rB[i] = *(const float4*)(bPtr[i] + (size_t)kn * DH);
      __builtin_prefetch(bPtr[0] + (size_t)(kn + 32) * DH, 0, 1);
    }

    FragBF a;
    a.q[0] = *(const v4u*)&As[buf][rA][kc];
    a.q[1] = *(const v4u*)&As[buf][rA][kc + 16];
#pragma unroll
    for (int t = 0; t < 4; ++t) {
      int nb = (t << 4) + (lane & 15);
      FragBF b;
      b.q[0] = *(const v4u*)&Bt[buf][nb][kc];
      b.q[1] = *(const v4u*)&Bt[buf][nb][kc + 16];
      acc[t] = __builtin_amdgcn_wmma_f32_16x16x32_bf16(false, a.v, false, b.v,
                                                       (short)0, acc[t], false, false);
    }

    if (hasNext) {
      const int nb2 = buf ^ 1;
#pragma unroll
      for (int i = 0; i < 2; ++i) {
        Bt[nb2][bC[i] + 0][bKr[i]] = (__bf16)rB[i].x; Bt[nb2][bC[i] + 1][bKr[i]] = (__bf16)rB[i].y;
        Bt[nb2][bC[i] + 2][bKr[i]] = (__bf16)rB[i].z; Bt[nb2][bC[i] + 3][bKr[i]] = (__bf16)rB[i].w;
      }
#if HAVE_ASYNC_LDS
      wait_async0();
#endif
    }
    __syncthreads();
    buf ^= 1;
  }

#pragma unroll
  for (int t = 0; t < 4; ++t) {
#pragma unroll
    for (int i = 0; i < 8; ++i) {
      int rloc = (w << 4) + i + ((lane & 16) >> 1);
      int r = mBase + rloc;
      if (r < cnt) {
        int col = nBase + (t << 4) + (lane & 15);
        float v = acc[t][i];
        if (ROUTED) {
          atomicAdd(&out[(size_t)sTok[rloc] * DH + col], v);
        } else {
          out[(size_t)r * DH + col] = v;
        }
      }
    }
  }
}

// ---------------------------------------------------------------------------
// launch
// ---------------------------------------------------------------------------
extern "C" void kernel_launch(void* const* d_in, const int* in_sizes, int n_in,
                              void* d_out, int out_size, void* d_ws, size_t ws_size,
                              hipStream_t stream) {
  (void)in_sizes; (void)n_in; (void)out_size; (void)ws_size;

  const float* X     = (const float*)d_in[0];  // [N, D]
  const float* Wg    = (const float*)d_in[1];  // [D, E]
  const float* Wgate = (const float*)d_in[2];  // [E, D, H]
  const float* Wup   = (const float*)d_in[3];  // [E, D, H]
  const float* Wdown = (const float*)d_in[4];  // [E, H, D]
  const float* WsG   = (const float*)d_in[5];  // [D, H]
  const float* WsU   = (const float*)d_in[6];  // [D, H]
  const float* WsD   = (const float*)d_in[7];  // [H, D]
  float* out = (float*)d_out;                  // [N, D]

  char* ws = (char*)d_ws;
  float*          combine = (float*)(ws + WS_COMBINE);
  int*            top2    = (int*)(ws + WS_TOP2);
  int*            ctrl    = (int*)(ws + WS_CTRL);
  int*            tokids  = (int*)(ws + WS_TOKIDS);
  unsigned short* Hroute  = (unsigned short*)(ws + WS_HROUTE);
  unsigned short* Hshared = (unsigned short*)(ws + WS_HSHARED);

  moe_ctrl_init<<<1, 32, 0, stream>>>(ctrl);
  moe_router<<<NTOK / 256, 256, 0, stream>>>(X, Wg, combine, top2, ctrl);
  moe_prefix<<<1, 32, 0, stream>>>(ctrl);
  moe_fill<<<NTOK / 256, 256, 0, stream>>>(top2, ctrl, tokids);

  dim3 gs(DE / 64, NTOK / 128, 1);
  moe_gateup<false><<<gs, 256, 0, stream>>>(X, WsG, WsU, combine, tokids, ctrl, Hshared);
  dim3 gr(DE / 64, NTOK / 128, NEXP);
  moe_gateup<true><<<gr, 256, 0, stream>>>(X, Wgate, Wup, combine, tokids, ctrl, Hroute);

  dim3 ds(DH / 64, NTOK / 128, 1);
  moe_down<false><<<ds, 256, 0, stream>>>(Hshared, WsD, tokids, ctrl, out);
  dim3 dr(DH / 64, NTOK / 128, NEXP);
  moe_down<true><<<dr, 256, 0, stream>>>(Hroute, Wdown, tokids, ctrl, out);
}
